// LogicEncoder_82867099009891
// MI455X (gfx1250) — matrix operand order repaired and checked by reading, not compiled, not run on hardware
//
#include <hip/hip_runtime.h>

typedef __attribute__((ext_vector_type(8)))  float  v8f;
typedef __attribute__((ext_vector_type(8)))  __bf16 v8bf;
typedef __attribute__((ext_vector_type(16))) __bf16 v16bf;

#define LD_   256
#define B_    128
#define NN_   1023
#define KCAT  768

union Frag { v16bf v; v8bf h[2]; };

__device__ __forceinline__ float sigf(float x){ return 1.0f/(1.0f+__expf(-x)); }

// Async copy 16B global -> LDS (CDNA5 GLOBAL_LOAD_ASYNC_TO_LDS_B128, ASYNCcnt-tracked).
__device__ __forceinline__ void async_cp16(unsigned int ldsAddr, const __bf16* gptr){
  asm volatile("global_load_async_to_lds_b128 %0, %1, off"
               :: "v"(ldsAddr), "v"(gptr) : "memory");
}
__device__ __forceinline__ void wait_async0(){
  asm volatile("s_wait_asynccnt 0x0" ::: "memory");
}
__device__ __forceinline__ unsigned int lds_addr(const void* p){
  return (unsigned int)(unsigned long long)p;   // low 32 bits = LDS offset
}

// ---------------- prep kernels ----------------
// x_embed [B][N][256] fp32 -> xb [node][batch][256] bf16
__global__ void cvt_x_kernel(const float* __restrict__ x, __bf16* __restrict__ xb){
  size_t i = (size_t)blockIdx.x*blockDim.x + threadIdx.x;
  size_t tot = (size_t)NN_*B_*LD_;
  if (i >= tot) return;
  int d = (int)(i % LD_);
  size_t t = i / LD_;
  int b = (int)(t % B_);
  int n = (int)(t / B_);
  xb[i] = (__bf16)x[((size_t)b*NN_ + n)*LD_ + d];
}

// leaf packed weight [256][512] : cols [Wcx | Wox]
__global__ void pack_wleaf_kernel(const float* __restrict__ Wcx, const float* __restrict__ Wox,
                                  __bf16* __restrict__ out){
  int i = blockIdx.x*blockDim.x + threadIdx.x;
  if (i >= 256*512) return;
  int j = i & 511, k = i >> 9;
  int g = j >> 8, jo = j & 255;
  out[i] = (__bf16)((g==0 ? Wcx : Wox)[k*256 + jo]);
}

// internal packed weight [768][1024] : row blocks x/hL/hR, col blocks i/f0/f1/u
__global__ void pack_wint_kernel(const float* __restrict__ Wix, const float* __restrict__ Wfx,
                                 const float* __restrict__ Wux, const float* __restrict__ Wi,
                                 const float* __restrict__ Wf, const float* __restrict__ Wu,
                                 __bf16* __restrict__ out){
  int i = blockIdx.x*blockDim.x + threadIdx.x;
  if (i >= KCAT*1024) return;
  int j = i & 1023, k = i >> 10;
  int g = j >> 8, jo = j & 255;       // gate, col within gate
  int seg = k >> 8, ko = k & 255;     // 0=x, 1=left child h, 2=right child h
  float v;
  if (seg == 0){
    const float* W = (g==0) ? Wix : (g==3) ? Wux : Wfx;
    v = W[ko*256 + jo];
  } else {
    int ch = seg - 1;                 // einsum child index j
    if (g==0)      v = Wi[(ch*256 + ko)*256 + jo];            // Wi[j]
    else if (g==3) v = Wu[(ch*256 + ko)*256 + jo];            // Wu[j]
    else           v = Wf[(((g-1)*2 + ch)*256 + ko)*256 + jo];// Wf[k=g-1][j]
  }
  out[i] = (__bf16)v;
}

__global__ void pack_wfin_kernel(const float* __restrict__ Woend, __bf16* __restrict__ out){
  int i = blockIdx.x*blockDim.x + threadIdx.x;
  if (i >= KCAT*256) return;
  out[i] = (__bf16)Woend[i];
}

__global__ void pack_bias_kernel(const float* __restrict__ bcx, const float* __restrict__ box_,
                                 const float* __restrict__ bix, const float* __restrict__ bfx,
                                 const float* __restrict__ bux, const float* __restrict__ bi,
                                 const float* __restrict__ bf, const float* __restrict__ bu,
                                 const float* __restrict__ boend,
                                 float* __restrict__ bleaf, float* __restrict__ bint,
                                 float* __restrict__ bfin){
  int j = blockIdx.x*blockDim.x + threadIdx.x;
  if (j < 512){
    int g = j >> 8, jo = j & 255;
    bleaf[j] = g ? box_[jo] : bcx[jo];
  }
  if (j < 256) bfin[j] = boend[j];
  if (j < 1024){
    int g = j >> 8, jo = j & 255;
    float v;
    if (g == 0)      v = bix[jo] + bi[jo] + bi[256 + jo];            // bix + bi.sum(0)
    else if (g == 3) v = bux[jo] + bu[jo] + bu[256 + jo];            // bux + bu.sum(0)
    else             v = bfx[jo] + bf[((g-1)*2    )*256 + jo]
                                 + bf[((g-1)*2 + 1)*256 + jo];       // bfx + bf.sum(1)[k]
    bint[j] = v;
  }
}

// A matrix for the head: [B][768] bf16 = [init | c_root | h_root]
__global__ void build_afin_kernel(const float* __restrict__ initEmb,
                                  const float* __restrict__ cAll,
                                  const __bf16* __restrict__ hAll,
                                  __bf16* __restrict__ afin){
  int i = blockIdx.x*blockDim.x + threadIdx.x;
  if (i >= B_*KCAT) return;
  int k = i % KCAT;
  int b = i / KCAT;
  float v;
  if (k < 256)      v = initEmb[b*256 + k];
  else if (k < 512) v = cAll[(size_t)b*LD_ + (k - 256)];   // node 0
  else              v = (float)hAll[(size_t)b*LD_ + (k - 512)];
  afin[i] = (__bf16)v;
}

// ---------------- fused WMMA GEMM + LSTM-cell epilogue ----------------
// Block = 8 wave32s. Tile: M=128 (= one node's batch), N=64 per gate, G gates.
// Double-buffered LDS staged by GLOBAL_LOAD_ASYNC_TO_LDS_B128 (loads of tile t+1
// overlap WMMAs of tile t).
// PHASE 0: leaf (G=2: c,o)   PHASE 1: internal (G=4: i,f0,f1,u)   PHASE 2: head (G=1, tanh)
template<int G, int KTOT, int PHASE>
__global__ void __launch_bounds__(256)
tree_gemm_kernel(const __bf16* __restrict__ Abase,   // xb (phase 0/1) or afin (phase 2)
                 const __bf16* __restrict__ hAll,    // child h (phase 1)
                 const __bf16* __restrict__ Wpk,     // packed weight [KTOT][G*256]
                 const float*  __restrict__ bias,    // packed bias  [G*256]
                 float*        __restrict__ cAll,
                 __bf16*       __restrict__ hOut,
                 float*        __restrict__ finOut,
                 int nodeBase)
{
  __shared__ alignas(16) __bf16 lsA[2][128][48];     // 128x32 tile, 16B-aligned rows
  __shared__ alignas(16) __bf16 lsB[2][G][32][72];   // per-gate 32x64 tile

  const int tid  = threadIdx.x;
  const int lane = tid & 31;
  const int wid  = tid >> 5;
  const int wm = (wid & 3) * 32;       // wave M origin
  const int wn = (wid >> 2) * 32;      // wave N origin (within 64-wide gate tile)
  const int colBase = blockIdx.x * 64; // gate-relative column tile
  const int node = nodeBase + blockIdx.y;

  v8f acc[G][2][2];
  #pragma unroll
  for (int g=0; g<G; ++g)
    #pragma unroll
    for (int mt=0; mt<2; ++mt)
      #pragma unroll
      for (int nt=0; nt<2; ++nt){
        v8f z = {0.f,0.f,0.f,0.f,0.f,0.f,0.f,0.f};
        acc[g][mt][nt] = z;
      }

  const int ar  = tid >> 1;            // A stage: row (each thread copies 32B of one row)
  const int ak  = (tid & 1) * 16;      // A stage: k offset
  const int bkr = tid >> 3;            // B stage: k row
  const int bnc = (tid & 7) * 8;       // B stage: col offset (16B per thread per gate)

  // per-thread global A address for k-tile k0
  auto a_gptr = [&](int k0) -> const __bf16* {
    int kg = k0 + ak;
    if constexpr (PHASE == 2){
      return Abase + (size_t)ar*KTOT + kg;
    } else if constexpr (PHASE == 0){
      return Abase + (size_t)node*(B_*LD_) + (size_t)ar*LD_ + kg;
    } else {
      int seg = kg >> 8, ko = kg & 255;            // 0:x  1:h(2n+1)  2:h(2n+2)
      const __bf16* base = (seg == 0)
          ? (Abase + (size_t)node*(B_*LD_))
          : (hAll  + (size_t)(2*node + seg)*(B_*LD_));
      return base + (size_t)ar*LD_ + ko;
    }
  };

  // issue all async copies for k-tile k0 into buffer `buf`
  auto issue_tile = [&](int k0, int buf){
    const __bf16* ap = a_gptr(k0);
    unsigned int la = lds_addr(&lsA[buf][ar][ak]);
    async_cp16(la,      ap);
    async_cp16(la + 16, ap + 8);
    const __bf16* wp = Wpk + (size_t)(k0 + bkr)*(G*256) + colBase + bnc;
    #pragma unroll
    for (int g=0; g<G; ++g)
      async_cp16(lds_addr(&lsB[buf][g][bkr][bnc]), wp + g*256);
    if (k0 + 64 < KTOT)
      __builtin_prefetch(wp + (size_t)64*(G*256), 0, 0);  // global_prefetch_b8, 2 tiles ahead
  };

  constexpr int T = KTOT / 32;
  issue_tile(0, 0);

  for (int t = 0; t < T; ++t){
    const int cur = t & 1;
    wait_async0();        // this wave's tile-t writes landed in LDS
    __syncthreads();      // all waves' tile-t writes visible; prior reads of !cur done
    if (t + 1 < T)
      issue_tile((t + 1) * 32, cur ^ 1);   // overlap with compute below

    // A fragment: lane<16 -> M=lane, K {0..7,16..23}; lane>=16 -> M=lane-16, K {8..15,24..31}
    const int lm = lane & 15;
    const int kq = (lane >> 4) << 3;
    Frag af[2];
    #pragma unroll
    for (int mt=0; mt<2; ++mt){
      int row = wm + mt*16 + lm;
      af[mt].h[0] = *(const v8bf*)&lsA[cur][row][kq];
      af[mt].h[1] = *(const v8bf*)&lsA[cur][row][16 + kq];
    }
    #pragma unroll
    for (int g=0; g<G; ++g){
      // B fragment: N = lane&15, K {0..7,16..23} (lane<16) / {8..15,24..31} (lane>=16)
      Frag bf2[2];
      #pragma unroll
      for (int nt=0; nt<2; ++nt){
        int col = wn + nt*16;
        #pragma unroll
        for (int s=0; s<8; ++s){
          bf2[nt].v[s]     = lsB[cur][g][kq + s][col + lm];
          bf2[nt].v[8 + s] = lsB[cur][g][16 + kq + s][col + lm];
        }
      }
      #pragma unroll
      for (int mt=0; mt<2; ++mt)
        #pragma unroll
        for (int nt=0; nt<2; ++nt)
          acc[g][mt][nt] = __builtin_amdgcn_wmma_f32_16x16x32_bf16(
              false, af[mt].v, false, bf2[nt].v, (short)0, acc[g][mt][nt], false, false);
    }
  }

  // epilogue: C layout -> row = e + 8*(lane>=16), col = lane&15 within a 16x16 tile
  const int rHalf = (lane >> 4) << 3;
  #pragma unroll
  for (int mt=0; mt<2; ++mt){
    #pragma unroll
    for (int nt=0; nt<2; ++nt){
      #pragma unroll
      for (int e=0; e<8; ++e){
        int r = wm + mt*16 + e + rHalf;                    // batch index
        int c = colBase + wn + nt*16 + (lane & 15);        // gate-relative column
        if constexpr (PHASE == 0){
          float cpre = acc[0][mt][nt][e] + bias[c];
          float opre = acc[1][mt][nt][e] + bias[256 + c];
          float hv = sigf(opre) * tanhf(cpre);
          size_t idx = (size_t)node*(B_*LD_) + (size_t)r*LD_ + c;
          cAll[idx] = cpre;
          hOut[idx] = (__bf16)hv;
        } else if constexpr (PHASE == 1){
          float ipre = acc[0][mt][nt][e] + bias[c];
          float f0   = acc[1][mt][nt][e] + bias[256 + c];  // reference: f NOT sigmoided
          float f1   = acc[2][mt][nt][e] + bias[512 + c];
          float upre = acc[3][mt][nt][e] + bias[768 + c];
          size_t rc = (size_t)r*LD_ + c;
          float cL = cAll[(size_t)(2*node + 1)*(B_*LD_) + rc];
          float cR = cAll[(size_t)(2*node + 2)*(B_*LD_) + rc];
          float cv = sigf(ipre)*tanhf(upre) + f0*cL + f1*cR;
          size_t idx = (size_t)node*(B_*LD_) + rc;
          cAll[idx] = cv;
          hOut[idx] = (__bf16)tanhf(cv);
        } else {
          finOut[(size_t)r*LD_ + c] = tanhf(acc[0][mt][nt][e] + bias[c]);
        }
      }
    }
  }
}

// ---------------- launcher ----------------
extern "C" void kernel_launch(void* const* d_in, const int* in_sizes, int n_in,
                              void* d_out, int out_size, void* d_ws, size_t ws_size,
                              hipStream_t stream){
  const float* x_embed  = (const float*)d_in[0];
  const float* init_emb = (const float*)d_in[1];
  const float* Wcx  = (const float*)d_in[2];
  const float* bcx  = (const float*)d_in[3];
  const float* Wox  = (const float*)d_in[4];
  const float* box_ = (const float*)d_in[5];
  const float* Wix  = (const float*)d_in[6];
  const float* bix  = (const float*)d_in[7];
  const float* Wfx  = (const float*)d_in[8];
  const float* bfx  = (const float*)d_in[9];
  const float* Wux  = (const float*)d_in[10];
  const float* bux  = (const float*)d_in[11];
  const float* Wi   = (const float*)d_in[12];
  const float* bi   = (const float*)d_in[13];
  const float* Wf   = (const float*)d_in[14];
  const float* bf   = (const float*)d_in[15];
  const float* Wu   = (const float*)d_in[16];
  const float* bu   = (const float*)d_in[17];
  const float* Woend= (const float*)d_in[18];
  const float* boend= (const float*)d_in[19];
  (void)in_sizes; (void)n_in; (void)out_size; (void)ws_size;

  char* ws = (char*)d_ws;
  size_t off = 0;
  auto carve = [&](size_t bytes)->char*{
    char* p = ws + off;
    off = (off + bytes + 255) & ~(size_t)255;
    return p;
  };
  __bf16* xb    = (__bf16*)carve((size_t)NN_*B_*LD_*2);   // 67 MB
  __bf16* hb    = (__bf16*)carve((size_t)NN_*B_*LD_*2);   // 67 MB
  float*  cAll  = (float*) carve((size_t)NN_*B_*LD_*4);   // 134 MB
  __bf16* wleaf = (__bf16*)carve((size_t)256*512*2);
  __bf16* wint  = (__bf16*)carve((size_t)KCAT*1024*2);
  __bf16* wfin  = (__bf16*)carve((size_t)KCAT*256*2);
  float*  bleaf = (float*) carve(512*4);
  float*  bint  = (float*) carve(1024*4);
  float*  bfin  = (float*) carve(256*4);
  __bf16* afin  = (__bf16*)carve((size_t)B_*KCAT*2);

  { // x -> bf16 [node][batch][d]
    size_t tot = (size_t)NN_*B_*LD_;
    cvt_x_kernel<<<dim3((unsigned)((tot + 255)/256)), 256, 0, stream>>>(x_embed, xb);
  }
  pack_wleaf_kernel<<<dim3((256*512 + 255)/256), 256, 0, stream>>>(Wcx, Wox, wleaf);
  pack_wint_kernel <<<dim3((KCAT*1024 + 255)/256), 256, 0, stream>>>(Wix, Wfx, Wux, Wi, Wf, Wu, wint);
  pack_wfin_kernel <<<dim3((KCAT*256 + 255)/256), 256, 0, stream>>>(Woend, wfin);
  pack_bias_kernel <<<dim3(4), 256, 0, stream>>>(bcx, box_, bix, bfx, bux, bi, bf, bu, boend,
                                                 bleaf, bint, bfin);

  // Leaves: nodes 511..1022, GEMM [B*512,256]x[256,512]
  tree_gemm_kernel<2, 256, 0><<<dim3(4, 512), 256, 0, stream>>>(
      xb, hb, wleaf, bleaf, cAll, hb, nullptr, 511);

  // Internal levels 8..0 (sequential dependence): GEMM [B*W,768]x[768,1024]
  for (int lvl = 8; lvl >= 0; --lvl){
    int base  = (1 << lvl) - 1;
    int width = 1 << lvl;
    tree_gemm_kernel<4, 768, 1><<<dim3(4, width), 256, 0, stream>>>(
        xb, hb, wint, bint, cAll, hb, nullptr, base);
  }

  // Head: tanh([init|c_root|h_root] @ Woend + boend)
  build_afin_kernel<<<dim3((B_*KCAT + 255)/256), 256, 0, stream>>>(init_emb, cAll, hb, afin);
  tree_gemm_kernel<1, 768, 2><<<dim3(4, 1), 256, 0, stream>>>(
      afin, hb, wfin, bfin, cAll, hb, (float*)d_out, 0);
}
